// SparseAttention_45956150067828
// MI455X (gfx1250) — matrix operand order repaired and checked by reading, not compiled, not run on hardware
//
#include <hip/hip_runtime.h>
#include <hip/hip_bf16.h>

typedef __attribute__((ext_vector_type(16))) _Float16 v16h;
typedef __attribute__((ext_vector_type(8)))  _Float16 v8h;
typedef __attribute__((ext_vector_type(8)))  float    v8f;

#define S_LEN 8192
#define DIMM  2048
#define NHEAD 32
#define DHEAD 64
#define N3    6144
#define NCPAD 160      // 4 mem + 128 compressed, padded to 160 (5 K-steps of 32)
#define SCALE_F 0.125f // 64^-0.5
#define NEGV  (-1.0e30f)

// ---- CDNA5 feature detection -------------------------------------------------
#if defined(__has_builtin)
#if __has_builtin(__builtin_amdgcn_tensor_load_to_lds) && __has_builtin(__builtin_amdgcn_s_wait_tensorcnt)
#define HAVE_TDM 1
#if __has_include(<hip/amd_detail/amd_gfx1250_TDM.h>)
#define TDM_6ARG 1
#endif
#endif
#endif

typedef __attribute__((ext_vector_type(4))) unsigned tdm_v4u;
typedef __attribute__((ext_vector_type(8))) int      tdm_v8i;
typedef __attribute__((ext_vector_type(4))) int      tdm_v4i;

// ---------------- WMMA helpers (CDNA5 16x16x32 f16 -> f32) ----------------

__device__ __forceinline__ v8f zero8(){
  v8f z = {0.f,0.f,0.f,0.f,0.f,0.f,0.f,0.f};
  return z;
}

__device__ __forceinline__ v8f wmma_f16(v16h a, v16h b, v8f c){
  // (neg_a, A, neg_b, B, c_mod, C, reuse_a, reuse_b)
  return __builtin_amdgcn_wmma_f32_16x16x32_f16(false, a, false, b, (short)0, c, false, false);
}

// Async DMA: global -> LDS, 16 bytes; INST_OFFSET applies to both addresses.
__device__ __forceinline__ void async_ld_b128(unsigned lds_addr, const void* gaddr){
  asm volatile("global_load_async_to_lds_b128 %0, %1, off" :: "v"(lds_addr), "v"(gaddr) : "memory");
}
__device__ __forceinline__ void async_ld_b128_off16(unsigned lds_addr, const void* gaddr){
  asm volatile("global_load_async_to_lds_b128 %0, %1, off offset:16" :: "v"(lds_addr), "v"(gaddr) : "memory");
}
__device__ __forceinline__ void wait_async0(){
  asm volatile("s_wait_asynccnt 0x0" ::: "memory");
}
__device__ __forceinline__ void wait_async2(){
  // async loads complete in order: <=2 outstanding means the older panel landed
  asm volatile("s_wait_asynccnt 0x2" ::: "memory");
}

#ifdef HAVE_TDM
// Tensor Data Mover: 2-D f16 tile, 64 elems wide x 64 rows, row stride in elements.
__device__ __forceinline__ void tdm_load_2d_f16_64x64(const _Float16* gsrc, unsigned lds_addr,
                                                      unsigned row_stride){
  unsigned long long ga = (unsigned long long)(size_t)gsrc;
  tdm_v4u g0;
  g0[0] = 1u;                                   // count=1, user descriptor
  g0[1] = lds_addr;                             // lds_addr (bytes)
  g0[2] = (unsigned)(ga & 0xFFFFFFFFu);         // global_addr[31:0]
  g0[3] = (unsigned)((ga >> 32) & 0x1FFFFFFu) | (2u << 30);  // addr[56:32] | type=2
  tdm_v8i g1;
  unsigned td0 = row_stride;                    // tensor_dim0 (elements)
  unsigned td1 = 1u << 20;                      // tensor_dim1 (rows, generous)
  g1[0] = (int)(1u << 16);                      // data_size=1 (2 bytes/elem)
  g1[1] = (int)((td0 & 0xFFFFu) << 16);         // tensor_dim0 lo16 in [31:16]
  g1[2] = (int)(((td0 >> 16) & 0xFFFFu) | ((td1 & 0xFFFFu) << 16));
  g1[3] = (int)(((td1 >> 16) & 0xFFFFu) | (64u << 16));  // tile_dim0 = 64
  g1[4] = 64;                                   // tile_dim1 = 64, tile_dim2 = 0
  g1[5] = (int)row_stride;                      // tensor_dim0_stride lo32
  g1[6] = 0;
  g1[7] = 0;
  tdm_v4i gz = {0,0,0,0};
#ifdef TDM_6ARG
  tdm_v8i gz8 = {0,0,0,0,0,0,0,0};
  __builtin_amdgcn_tensor_load_to_lds(g0, g1, gz, gz, gz8, 0);
#else
  __builtin_amdgcn_tensor_load_to_lds(g0, g1, gz, gz, 0);
#endif
}
#endif

// A fragment: 16x32 f16, row-major source with leading dim lda.
// Lane L (g=L>>4, m=L&15): elements 0..7 = K k0+8g.. , elements 8..15 = K k0+16+8g..
__device__ __forceinline__ v16h load_a_frag(const _Float16* A, int lda, int m0, int k0, int lane){
  int g = lane >> 4;
  int m = m0 + (lane & 15);
  const _Float16* base = A + (size_t)m * lda;
  v8h lo = *(const v8h*)(base + k0 + 8*g);
  v8h hi = *(const v8h*)(base + k0 + 16 + 8*g);
  v16h a;
  #pragma unroll
  for (int e=0;e<8;e++){ a[e] = lo[e]; a[e+8] = hi[e]; }
  return a;
}

// B fragment: 32x16 f16 from B-transposed layout Bt[N][K] (K contiguous).
// Lane L: n = n0 + (L&15), elements 0..7 = K k0+8g.., elements 8..15 = K k0+16+8g..
__device__ __forceinline__ v16h load_b_frag(const _Float16* Bt, int ldb, int n0, int k0, int lane){
  int g = lane >> 4;
  int n = n0 + (lane & 15);
  const _Float16* base = Bt + (size_t)n * ldb;
  v8h lo = *(const v8h*)(base + k0 + 8*g);
  v8h hi = *(const v8h*)(base + k0 + 16 + 8*g);
  v16h b;
  #pragma unroll
  for (int e=0;e<8;e++){ b[e] = lo[e]; b[e+8] = hi[e]; }
  return b;
}

// Reductions across the 16-lane half-wave (rows of a C tile live in one half).
__device__ __forceinline__ float redmax16(float v){
  v = fmaxf(v, __shfl_xor(v, 1, 32));
  v = fmaxf(v, __shfl_xor(v, 2, 32));
  v = fmaxf(v, __shfl_xor(v, 4, 32));
  v = fmaxf(v, __shfl_xor(v, 8, 32));
  return v;
}
__device__ __forceinline__ float redsum16(float v){
  v += __shfl_xor(v, 1, 32);
  v += __shfl_xor(v, 2, 32);
  v += __shfl_xor(v, 4, 32);
  v += __shfl_xor(v, 8, 32);
  return v;
}

// ---------------- K0: RMSNorm + f32->f16 ----------------

__global__ __launch_bounds__(256) void rms_kernel(const float* __restrict__ inp,
                                                  const float* __restrict__ w,
                                                  _Float16* __restrict__ xh){
  int s = blockIdx.x;
  const float* row = inp + (size_t)s * DIMM;
  float acc = 0.f;
  for (int i = threadIdx.x; i < DIMM; i += 256){ float v = row[i]; acc += v*v; }
  #pragma unroll
  for (int m=1;m<32;m<<=1) acc += __shfl_xor(acc, m, 32);
  __shared__ float red[8];
  int wave = threadIdx.x >> 5, lane = threadIdx.x & 31;
  if (lane == 0) red[wave] = acc;
  __syncthreads();
  float tot = red[0]+red[1]+red[2]+red[3]+red[4]+red[5]+red[6]+red[7];
  float inv = rsqrtf(tot / (float)DIMM + 1.1920929e-07f);
  _Float16* orow = xh + (size_t)s * DIMM;
  for (int i = threadIdx.x; i < DIMM; i += 256)
    orow[i] = (_Float16)(row[i] * inv * w[i]);
}

// ---------------- weight conversion kernels ----------------

// out[n*K + k] = (f16) in[k*N + n], rows n >= N zero-padded (Npad total rows)
__global__ __launch_bounds__(256) void transpose_cvt_kernel(const float* __restrict__ in,
                                                            _Float16* __restrict__ out,
                                                            int K, int N, int Npad){
  size_t idx = (size_t)blockIdx.x * 256 + threadIdx.x;
  if (idx >= (size_t)Npad * K) return;
  int n = (int)(idx / K);
  int k = (int)(idx - (size_t)n * K);
  out[idx] = (n < N) ? (_Float16)in[(size_t)k * N + n] : (_Float16)0.f;
}

// conv weight [h][o][i][t] -> [h*64+o][t*64+i]  (Bt layout, K = t*64+i)
__global__ __launch_bounds__(256) void convw_cvt_kernel(const float* __restrict__ in,
                                                        _Float16* __restrict__ out){
  size_t idx = (size_t)blockIdx.x * 256 + threadIdx.x;   // 32*64*4096 total
  int ho  = (int)(idx >> 12);
  int rem = (int)(idx & 4095);
  int t = rem >> 6, i = rem & 63;
  out[idx] = (_Float16)in[((size_t)ho * 64 + i) * 64 + t];
}

__global__ __launch_bounds__(256) void cvt_kernel(const float* __restrict__ in,
                                                  _Float16* __restrict__ out, int n){
  int idx = blockIdx.x * 256 + threadIdx.x;
  if (idx < n) out[idx] = (_Float16)in[idx];
}

// ---------------- templated WMMA GEMM: C[MxN] = A[MxK] * Bt[NxK]^T ----------------
// block tile: M=128 (wave w -> m-tile w), N=64 (4 n-tiles per wave).
// B double-buffered in LDS via async global->LDS DMA pipelined against WMMA.
// MODE 0: scatter qkv (f16) head-major (+ transposed v). MODE 1: sigmoid gates (f32).
// MODE 2: plain f32 store (final projection).

template<int MODE>
__global__ __launch_bounds__(256) void wmma_gemm_kernel(
    const _Float16* __restrict__ A, const _Float16* __restrict__ Bt,
    int lda, int ldb, int K,
    _Float16* __restrict__ oQ, _Float16* __restrict__ oK,
    _Float16* __restrict__ oV, _Float16* __restrict__ oVT,
    float* __restrict__ oF, const float* __restrict__ bias)
{
  __shared__ _Float16 sB[2][64 * 64];   // 2 x 8KB K-panels (ping-pong)
  int n0 = blockIdx.x * 64;
  int wave = threadIdx.x >> 5, lane = threadIdx.x & 31;
  int lane15 = lane & 15, g = lane >> 4;
  int m0 = blockIdx.y * 128 + wave * 16;
  v8f c0 = zero8(), c1 = zero8(), c2 = zero8(), c3 = zero8();
  int sn = threadIdx.x >> 2, sc = threadIdx.x & 3;          // row 0..63, 32B segment 0..3
  const _Float16* bsrc = Bt + (size_t)(n0 + sn) * ldb + 16 * sc;
  unsigned lds_dst[2];
  lds_dst[0] = (unsigned)(size_t)(sB[0] + sn * 64 + 16 * sc);
  lds_dst[1] = (unsigned)(size_t)(sB[1] + sn * 64 + 16 * sc);
  int nk = K >> 6;
  // prologue: panel 0 in flight
  async_ld_b128(lds_dst[0], bsrc);
  async_ld_b128_off16(lds_dst[0], bsrc);
  for (int i = 0; i < nk; i++){
    int k0 = i << 6;
    __syncthreads();                       // all waves done reading buf[(i+1)&1] (2 panels ago)
    if (i + 1 < nk){
      async_ld_b128(lds_dst[(i + 1) & 1], bsrc + k0 + 64);
      async_ld_b128_off16(lds_dst[(i + 1) & 1], bsrc + k0 + 64);
      __builtin_prefetch(bsrc + k0 + 128, 0, 1);
      wait_async2();                       // panel i resident (in-order completion)
    } else {
      wait_async0();
    }
    __syncthreads();                       // panel i visible block-wide
    const _Float16* sBc = sB[i & 1];
    v16h a0 = load_a_frag(A, lda, m0, k0, lane);
    v16h a1 = load_a_frag(A, lda, m0, k0 + 32, lane);
    c0 = wmma_f16(a0, load_b_frag(sBc, 64,  0,  0, lane), c0);
    c1 = wmma_f16(a0, load_b_frag(sBc, 64, 16,  0, lane), c1);
    c2 = wmma_f16(a0, load_b_frag(sBc, 64, 32,  0, lane), c2);
    c3 = wmma_f16(a0, load_b_frag(sBc, 64, 48,  0, lane), c3);
    c0 = wmma_f16(a1, load_b_frag(sBc, 64,  0, 32, lane), c0);
    c1 = wmma_f16(a1, load_b_frag(sBc, 64, 16, 32, lane), c1);
    c2 = wmma_f16(a1, load_b_frag(sBc, 64, 32, 32, lane), c2);
    c3 = wmma_f16(a1, load_b_frag(sBc, 64, 48, 32, lane), c3);
  }
  v8f cc[4] = {c0, c1, c2, c3};
  #pragma unroll
  for (int nt = 0; nt < 4; nt++){
    int col = n0 + nt * 16 + lane15;
    if (MODE == 0){
      int which = col >> 11;          // 0=q 1=k 2=v
      int h = (col >> 6) & 31;
      int d = col & 63;
      #pragma unroll
      for (int r = 0; r < 8; r++){
        int srow = m0 + r + 8 * g;
        _Float16 val = (_Float16)cc[nt][r];
        size_t off = ((size_t)h * S_LEN + srow) * DHEAD + d;
        if (which == 0) oQ[off] = val;
        else if (which == 1) oK[off] = val;
        else { oV[off] = val; oVT[((size_t)h * DHEAD + d) * S_LEN + srow] = val; }
      }
    } else if (MODE == 1){
      if (col < 96){
        float b = bias[col];
        #pragma unroll
        for (int r = 0; r < 8; r++){
          int srow = m0 + r + 8 * g;
          float t = cc[nt][r] + b;
          oF[(size_t)srow * 96 + col] = 1.f / (1.f + __expf(-t));
        }
      }
    } else {
      #pragma unroll
      for (int r = 0; r < 8; r++){
        int srow = m0 + r + 8 * g;
        oF[(size_t)srow * DIMM + col] = cc[nt][r];
      }
    }
  }
}

// ---------------- compression: per head C[128x64] = (kv+pos)[128x4096] @ cwT ----------------

__device__ __forceinline__ v16h load_a_frag_comp(const _Float16* kv, const _Float16* pos,
                                                 int h, int jbase, int k0, int lane){
  int g = lane >> 4;
  int j = jbase + (lane & 15);
  int kk0 = k0 + 8 * g;
  int kk1 = k0 + 16 + 8 * g;
  int t0 = kk0 >> 6, i0 = kk0 & 63;
  int t1 = kk1 >> 6, i1 = kk1 & 63;
  v8h s0 = *(const v8h*)(kv  + ((size_t)h * S_LEN + j * 64 + t0) * 64 + i0);
  v8h p0 = *(const v8h*)(pos + ((size_t)h * 64 + t0) * 64 + i0);
  v8h s1 = *(const v8h*)(kv  + ((size_t)h * S_LEN + j * 64 + t1) * 64 + i1);
  v8h p1 = *(const v8h*)(pos + ((size_t)h * 64 + t1) * 64 + i1);
  v16h a;
  #pragma unroll
  for (int e = 0; e < 8; e++){
    a[e]     = (_Float16)(s0[e] + p0[e]);
    a[e + 8] = (_Float16)(s1[e] + p1[e]);
  }
  return a;
}

__global__ __launch_bounds__(256) void compress_kernel(
    const _Float16* __restrict__ kvh, const _Float16* __restrict__ posh,
    const _Float16* __restrict__ cwT, const float* __restrict__ bias,
    _Float16* __restrict__ out, int transposed)
{
  __shared__ _Float16 sB[2][64 * 64];   // 2 x 8KB K-panels of conv weights
  int h = blockIdx.x;
  int wave = threadIdx.x >> 5, lane = threadIdx.x & 31;
  int lane15 = lane & 15, g = lane >> 4;
  int jbase = wave * 16;
  v8f c0 = zero8(), c1 = zero8(), c2 = zero8(), c3 = zero8();
  const _Float16* cw = cwT + (size_t)h * 64 * 4096;
  const int nk = 4096 >> 6;
#ifdef HAVE_TDM
  unsigned lds_base[2];
  lds_base[0] = (unsigned)(size_t)(&sB[0][0]);
  lds_base[1] = (unsigned)(size_t)(&sB[1][0]);
  if (wave == 0) tdm_load_2d_f16_64x64(cw, lds_base[0], 4096u);
#else
  int sn = threadIdx.x >> 2, sc = threadIdx.x & 3;
  const _Float16* bsrc = cw + (size_t)sn * 4096 + 16 * sc;
  unsigned lds_dst[2];
  lds_dst[0] = (unsigned)(size_t)(sB[0] + sn * 64 + 16 * sc);
  lds_dst[1] = (unsigned)(size_t)(sB[1] + sn * 64 + 16 * sc);
  async_ld_b128(lds_dst[0], bsrc);
  async_ld_b128_off16(lds_dst[0], bsrc);
#endif
  for (int i = 0; i < nk; i++){
    int k0 = i << 6;
    __syncthreads();                       // safe to overwrite buf[(i+1)&1]
#ifdef HAVE_TDM
    if (i + 1 < nk){
      if (wave == 0) tdm_load_2d_f16_64x64(cw + k0 + 64, lds_base[(i + 1) & 1], 4096u);
      __builtin_amdgcn_s_wait_tensorcnt(1);   // panel i done (TDM in-order per wave)
    } else {
      __builtin_amdgcn_s_wait_tensorcnt(0);
    }
#else
    if (i + 1 < nk){
      async_ld_b128(lds_dst[(i + 1) & 1], bsrc + k0 + 64);
      async_ld_b128_off16(lds_dst[(i + 1) & 1], bsrc + k0 + 64);
      wait_async2();
    } else {
      wait_async0();
    }
#endif
    __syncthreads();                       // panel i visible block-wide
    const _Float16* sBc = sB[i & 1];
    v16h a0 = load_a_frag_comp(kvh, posh, h, jbase, k0, lane);
    v16h a1 = load_a_frag_comp(kvh, posh, h, jbase, k0 + 32, lane);
    c0 = wmma_f16(a0, load_b_frag(sBc, 64,  0,  0, lane), c0);
    c1 = wmma_f16(a0, load_b_frag(sBc, 64, 16,  0, lane), c1);
    c2 = wmma_f16(a0, load_b_frag(sBc, 64, 32,  0, lane), c2);
    c3 = wmma_f16(a0, load_b_frag(sBc, 64, 48,  0, lane), c3);
    c0 = wmma_f16(a1, load_b_frag(sBc, 64,  0, 32, lane), c0);
    c1 = wmma_f16(a1, load_b_frag(sBc, 64, 16, 32, lane), c1);
    c2 = wmma_f16(a1, load_b_frag(sBc, 64, 32, 32, lane), c2);
    c3 = wmma_f16(a1, load_b_frag(sBc, 64, 48, 32, lane), c3);
  }
  v8f cc[4] = {c0, c1, c2, c3};
  #pragma unroll
  for (int nt = 0; nt < 4; nt++){
    int o = nt * 16 + lane15;
    float bo = bias[h * 64 + o];
    #pragma unroll
    for (int r = 0; r < 8; r++){
      int j = jbase + r + 8 * g;
      float val = cc[nt][r] + bo;
      if (!transposed) out[((size_t)h * NCPAD + 4 + j) * DHEAD + o] = (_Float16)val;      // ck[h][j][d]
      else             out[((size_t)h * DHEAD + o) * NCPAD + 4 + j] = (_Float16)val;      // cvT[h][d][j]
    }
  }
}

// mem rows (j<4) + zero padding (j>=132) of ck / cvT
__global__ __launch_bounds__(256) void initmem_kernel(const float* __restrict__ mem,
                                                      _Float16* __restrict__ ck,
                                                      _Float16* __restrict__ cvT){
  int idx = blockIdx.x * 256 + threadIdx.x;   // 32*160*64
  int h = idx / (NCPAD * DHEAD);
  int r = (idx / DHEAD) % NCPAD;
  int d = idx % DHEAD;
  size_t tidx = ((size_t)h * DHEAD + d) * NCPAD + r;
  if (r < 4){
    ck[idx]   = (_Float16)mem[(((size_t)h) * 4 + r) * DHEAD + d];
    cvT[tidx] = (_Float16)mem[(((size_t)NHEAD + h) * 4 + r) * DHEAD + d];
  } else if (r >= 132){
    ck[idx]   = (_Float16)0.f;
    cvT[tidx] = (_Float16)0.f;
  }
}

// ---------------- compressed attention ----------------

__global__ __launch_bounds__(256) void cattn_kernel(
    const _Float16* __restrict__ qh, const _Float16* __restrict__ ck,
    const _Float16* __restrict__ cvT, _Float16* __restrict__ comp)
{
  __shared__ _Float16 attnS[8][16 * NCPAD];   // 40KB
  int h = blockIdx.y;
  int wave = threadIdx.x >> 5, lane = threadIdx.x & 31;
  int lane15 = lane & 15, g = lane >> 4;
  int m0 = blockIdx.x * 128 + wave * 16;
  const _Float16* Aq  = qh  + (size_t)h * S_LEN * DHEAD;
  const _Float16* Bck = ck  + (size_t)h * NCPAD * DHEAD;
  const _Float16* Bcv = cvT + (size_t)h * DHEAD * NCPAD;

  v16h a0 = load_a_frag(Aq, DHEAD, m0, 0, lane);
  v16h a1 = load_a_frag(Aq, DHEAD, m0, 32, lane);
  v8f sim[10];
  #pragma unroll
  for (int nt = 0; nt < 10; nt++){
    v8f acc = zero8();
    acc = wmma_f16(a0, load_b_frag(Bck, DHEAD, nt * 16,  0, lane), acc);
    acc = wmma_f16(a1, load_b_frag(Bck, DHEAD, nt * 16, 32, lane), acc);
    sim[nt] = acc;
  }
  // scale + mask (faithful to reference: mask where ck_seq[j] < s, plus padding)
  int jj = lane15;
  #pragma unroll
  for (int nt = 0; nt < 10; nt++){
    int j = nt * 16 + jj;
    int ckseq = (j < 4) ? -1 : (j - 3) * 64 - 1;
    #pragma unroll
    for (int r = 0; r < 8; r++){
      int s = m0 + r + 8 * g;
      float v = sim[nt][r] * SCALE_F;
      bool masked = (j >= 132) || (ckseq < s);
      sim[nt][r] = masked ? NEGV : v;
    }
  }
  // row softmax (row lives in one 16-lane group x 10 tiles)
  #pragma unroll
  for (int r = 0; r < 8; r++){
    float mx = NEGV;
    #pragma unroll
    for (int nt = 0; nt < 10; nt++) mx = fmaxf(mx, sim[nt][r]);
    mx = redmax16(mx);
    float sum = 0.f;
    #pragma unroll
    for (int nt = 0; nt < 10; nt++){
      float e = (sim[nt][r] > 0.5f * NEGV) ? __expf(sim[nt][r] - mx) : 0.f;
      sim[nt][r] = e;
      sum += e;
    }
    sum = redsum16(sum);
    float inv = 1.f / sum;
    #pragma unroll
    for (int nt = 0; nt < 10; nt++)
      attnS[wave][(r + 8 * g) * NCPAD + nt * 16 + lane15] = (_Float16)(sim[nt][r] * inv);
  }
  // out = attn @ cv
  const _Float16* As = &attnS[wave][0];
  v8f o0 = zero8(), o1 = zero8(), o2 = zero8(), o3 = zero8();
  #pragma unroll
  for (int ks = 0; ks < 5; ks++){
    v16h aA = load_a_frag(As, NCPAD, 0, ks * 32, lane);
    o0 = wmma_f16(aA, load_b_frag(Bcv, NCPAD,  0, ks * 32, lane), o0);
    o1 = wmma_f16(aA, load_b_frag(Bcv, NCPAD, 16, ks * 32, lane), o1);
    o2 = wmma_f16(aA, load_b_frag(Bcv, NCPAD, 32, ks * 32, lane), o2);
    o3 = wmma_f16(aA, load_b_frag(Bcv, NCPAD, 48, ks * 32, lane), o3);
  }
  v8f oo[4] = {o0, o1, o2, o3};
  #pragma unroll
  for (int dt = 0; dt < 4; dt++){
    int d = dt * 16 + lane15;
    #pragma unroll
    for (int r = 0; r < 8; r++){
      int s = m0 + r + 8 * g;
      comp[(size_t)s * DIMM + h * DHEAD + d] = (_Float16)oo[dt][r];
    }
  }
}

// ---------------- local sliding-window attention (flash-style) ----------------

__global__ __launch_bounds__(256) void lattn_kernel(
    const _Float16* __restrict__ qh, const _Float16* __restrict__ kh,
    const _Float16* __restrict__ vT, _Float16* __restrict__ localout)
{
  __shared__ _Float16 attnS[8][16 * 32];   // 8KB
  int h = blockIdx.y, nb = blockIdx.x;
  int wave = threadIdx.x >> 5, lane = threadIdx.x & 31;
  int lane15 = lane & 15, g = lane >> 4;
  int m0 = nb * 128 + wave * 16;
  const _Float16* Aq = qh + (size_t)h * S_LEN * DHEAD;
  const _Float16* Bk = kh + (size_t)h * S_LEN * DHEAD;
  const _Float16* Bv = vT + (size_t)h * DHEAD * S_LEN;

  v16h a0 = load_a_frag(Aq, DHEAD, m0, 0, lane);
  v16h a1 = load_a_frag(Aq, DHEAD, m0, 32, lane);
  v8f o[4];
  #pragma unroll
  for (int i = 0; i < 4; i++) o[i] = zero8();
  float runmax[8], runsum[8];
  #pragma unroll
  for (int r = 0; r < 8; r++){ runmax[r] = NEGV; runsum[r] = 0.f; }

  for (int c = 0; c < 8; c++){
    int kb = (nb - 1) * 128 + c * 32;     // key chunk base (32 keys)
    if (kb < 0) continue;                  // entirely before position 0 -> fully masked
    v8f sA = zero8(), sB2 = zero8();
    sA  = wmma_f16(a0, load_b_frag(Bk, DHEAD, kb,      0, lane), sA);
    sA  = wmma_f16(a1, load_b_frag(Bk, DHEAD, kb,     32, lane), sA);
    sB2 = wmma_f16(a0, load_b_frag(Bk, DHEAD, kb + 16,  0, lane), sB2);
    sB2 = wmma_f16(a1, load_b_frag(Bk, DHEAD, kb + 16, 32, lane), sB2);
    int jA = kb + lane15, jB = kb + 16 + lane15;
    #pragma unroll
    for (int r = 0; r < 8; r++){
      int i = m0 + r + 8 * g;
      int dA = i - jA, dB = i - jB;
      bool mA = (dA < 0) || (dA > 128);
      bool mB = (dB < 0) || (dB > 128);
      float vA = mA ? NEGV : sA[r]  * SCALE_F;
      float vB = mB ? NEGV : sB2[r] * SCALE_F;
      float tmx = redmax16(fmaxf(vA, vB));
      float nmx = fmaxf(runmax[r], tmx);
      float fac = __expf(runmax[r] - nmx);
      runmax[r] = nmx;
      float pA = mA ? 0.f : __expf(vA - nmx);
      float pB = mB ? 0.f : __expf(vB - nmx);
      runsum[r] = runsum[r] * fac + redsum16(pA + pB);
      #pragma unroll
      for (int dt = 0; dt < 4; dt++) o[dt][r] *= fac;
      attnS[wave][(r + 8 * g) * 32 + lane15]      = (_Float16)pA;
      attnS[wave][(r + 8 * g) * 32 + 16 + lane15] = (_Float16)pB;
    }
    v16h aA = load_a_frag(&attnS[wave][0], 32, 0, 0, lane);
    #pragma unroll
    for (int dt = 0; dt < 4; dt++)
      o[dt] = wmma_f16(aA, load_b_frag(Bv, S_LEN, dt * 16, kb, lane), o[dt]);
  }
  #pragma unroll
  for (int dt = 0; dt < 4; dt++){
    int d = dt * 16 + lane15;
    #pragma unroll
    for (int r = 0; r < 8; r++){
      int s = m0 + r + 8 * g;
      localout[(size_t)s * DIMM + h * DHEAD + d] = (_Float16)(o[dt][r] / runsum[r]);
    }
  }
}

// ---------------- gated combine ----------------

__global__ __launch_bounds__(256) void combine_kernel(const float* __restrict__ gates,
    const _Float16* __restrict__ localb, const _Float16* __restrict__ compb,
    _Float16* __restrict__ combh){
  size_t idx = (size_t)blockIdx.x * 256 + threadIdx.x;   // 8192*2048
  int col = (int)(idx & 2047);
  int s   = (int)(idx >> 11);
  int h = col >> 6;
  const float* gp = gates + (size_t)s * 96 + h * 3;
  float val = (gp[0] + gp[1]) * (float)localb[idx] + gp[2] * (float)compb[idx];
  combh[idx] = (_Float16)val;
}

// ---------------- host launcher ----------------

extern "C" void kernel_launch(void* const* d_in, const int* in_sizes, int n_in,
                              void* d_out, int out_size, void* d_ws, size_t ws_size,
                              hipStream_t stream){
  (void)in_sizes; (void)n_in; (void)out_size; (void)ws_size;
  const float* inp    = (const float*)d_in[0];
  const float* rms_w  = (const float*)d_in[1];
  const float* w_qkv  = (const float*)d_in[2];
  const float* k_pos  = (const float*)d_in[3];
  const float* v_pos  = (const float*)d_in[4];
  const float* kc_w   = (const float*)d_in[5];
  const float* kc_b   = (const float*)d_in[6];
  const float* vc_w   = (const float*)d_in[7];
  const float* vc_b   = (const float*)d_in[8];
  const float* mem_kv = (const float*)d_in[9];
  const float* w_comb = (const float*)d_in[10];
  const float* b_comb = (const float*)d_in[11];
  const float* w_out  = (const float*)d_in[12];

  char* base = (char*)d_ws;
  size_t off = 0;
  auto alloc = [&](size_t bytes)->void*{
    void* p = base + off;
    off = (off + bytes + 255) & ~(size_t)255;
    return p;
  };
  _Float16* xh     = (_Float16*)alloc((size_t)S_LEN * DIMM * 2);
  _Float16* wqkvT  = (_Float16*)alloc((size_t)N3 * DIMM * 2);
  _Float16* woutT  = (_Float16*)alloc((size_t)DIMM * DIMM * 2);
  _Float16* wcombT = (_Float16*)alloc((size_t)128 * DIMM * 2);
  _Float16* kcwT   = (_Float16*)alloc((size_t)NHEAD * 64 * 4096 * 2);
  _Float16* vcwT   = (_Float16*)alloc((size_t)NHEAD * 64 * 4096 * 2);
  _Float16* kposh  = (_Float16*)alloc((size_t)NHEAD * 64 * 64 * 2);
  _Float16* vposh  = (_Float16*)alloc((size_t)NHEAD * 64 * 64 * 2);
  _Float16* qh     = (_Float16*)alloc((size_t)NHEAD * S_LEN * DHEAD * 2);
  _Float16* kh     = (_Float16*)alloc((size_t)NHEAD * S_LEN * DHEAD * 2);
  _Float16* vh     = (_Float16*)alloc((size_t)NHEAD * S_LEN * DHEAD * 2);
  _Float16* vhT    = (_Float16*)alloc((size_t)NHEAD * S_LEN * DHEAD * 2);
  _Float16* ckbuf  = (_Float16*)alloc((size_t)NHEAD * NCPAD * DHEAD * 2);
  _Float16* cvTb   = (_Float16*)alloc((size_t)NHEAD * DHEAD * NCPAD * 2);
  float*    gates  = (float*)   alloc((size_t)S_LEN * 96 * 4);
  _Float16* compb  = (_Float16*)alloc((size_t)S_LEN * DIMM * 2);
  _Float16* localb = (_Float16*)alloc((size_t)S_LEN * DIMM * 2);
  _Float16* combh  = (_Float16*)alloc((size_t)S_LEN * DIMM * 2);

  rms_kernel<<<S_LEN, 256, 0, stream>>>(inp, rms_w, xh);

  transpose_cvt_kernel<<<(unsigned)(((size_t)N3 * DIMM + 255) / 256), 256, 0, stream>>>(
      w_qkv, wqkvT, DIMM, N3, N3);
  transpose_cvt_kernel<<<(unsigned)(((size_t)DIMM * DIMM + 255) / 256), 256, 0, stream>>>(
      w_out, woutT, DIMM, DIMM, DIMM);
  transpose_cvt_kernel<<<(unsigned)(((size_t)128 * DIMM + 255) / 256), 256, 0, stream>>>(
      w_comb, wcombT, DIMM, 96, 128);
  convw_cvt_kernel<<<32768, 256, 0, stream>>>(kc_w, kcwT);
  convw_cvt_kernel<<<32768, 256, 0, stream>>>(vc_w, vcwT);
  cvt_kernel<<<512, 256, 0, stream>>>(k_pos, kposh, NHEAD * 64 * 64);
  cvt_kernel<<<512, 256, 0, stream>>>(v_pos, vposh, NHEAD * 64 * 64);

  // QKV projection (scatter to head-major f16, plus transposed V)
  wmma_gemm_kernel<0><<<dim3(N3 / 64, S_LEN / 128), 256, 0, stream>>>(
      xh, wqkvT, DIMM, DIMM, DIMM, qh, kh, vh, vhT, nullptr, nullptr);
  // gate projection + sigmoid
  wmma_gemm_kernel<1><<<dim3(2, S_LEN / 128), 256, 0, stream>>>(
      xh, wcombT, DIMM, DIMM, DIMM, nullptr, nullptr, nullptr, nullptr, gates, b_comb);

  initmem_kernel<<<(NHEAD * NCPAD * DHEAD) / 256, 256, 0, stream>>>(mem_kv, ckbuf, cvTb);
  compress_kernel<<<NHEAD, 256, 0, stream>>>(kh, kposh, kcwT, kc_b, ckbuf, 0);
  compress_kernel<<<NHEAD, 256, 0, stream>>>(vh, vposh, vcwT, vc_b, cvTb, 1);

  cattn_kernel<<<dim3(S_LEN / 128, NHEAD), 256, 0, stream>>>(qh, ckbuf, cvTb, compb);
  lattn_kernel<<<dim3(S_LEN / 128, NHEAD), 256, 0, stream>>>(qh, kh, vhT, localb);

  combine_kernel<<<(unsigned)(((size_t)S_LEN * DIMM) / 256), 256, 0, stream>>>(
      gates, localb, compb, combh);

  // output projection -> f32 d_out
  wmma_gemm_kernel<2><<<dim3(DIMM / 64, S_LEN / 128), 256, 0, stream>>>(
      combh, woutT, DIMM, DIMM, DIMM, nullptr, nullptr, nullptr, nullptr,
      (float*)d_out, nullptr);
}
